// Encoder_27616639713913
// MI455X (gfx1250) — matrix operand order repaired and checked by reading, not compiled, not run on hardware
//
#include <hip/hip_runtime.h>
#include <math.h>

#define N_NODES 50000
#define N_EDGES 800000
#define C 256
#define NL 4
#define CAT ((2*NL+1)*C)   /* 2304 */
#define HID 128
#define NOUT 64

typedef __attribute__((ext_vector_type(8)))  __bf16 v8bf;
typedef __attribute__((ext_vector_type(16))) __bf16 v16bf;
typedef __attribute__((ext_vector_type(8)))  float  v8f;

__device__ __forceinline__ __bf16 f2bf(float f) {
  unsigned u = __float_as_uint(f);
  u = (u + 0x7FFFu + ((u >> 16) & 1u)) >> 16;
  unsigned short s = (unsigned short)u;
  __bf16 r;
  __builtin_memcpy(&r, &s, 2);
  return r;
}
__device__ __forceinline__ float bf2f(__bf16 b) {
  unsigned short s;
  __builtin_memcpy(&s, &b, 2);
  return __uint_as_float((unsigned)s << 16);
}

// A fragment (16x32 bf16): per-lane two 8-elem groups at +0 and +16 along K
__device__ __forceinline__ v16bf ld_afrag(const __bf16* p) {
  v8bf lo = *(const v8bf*)p;
  v8bf hi = *(const v8bf*)(p + 16);
  return __builtin_shufflevector(lo, hi, 0,1,2,3,4,5,6,7,8,9,10,11,12,13,14,15);
}
// B fragment (32x16 bf16): per-lane two 8-elem groups at +0 and +16 along K
__device__ __forceinline__ v16bf ld_bfrag(const __bf16* p) {
  v8bf lo = *(const v8bf*)p;
  v8bf hi = *(const v8bf*)(p + 16);
  return __builtin_shufflevector(lo, hi, 0,1,2,3,4,5,6,7,8,9,10,11,12,13,14,15);
}

// ---------------- utility kernels ----------------

__global__ void k_zero(float* p, size_t n) {
  size_t stride = (size_t)gridDim.x * blockDim.x;
  for (size_t i = (size_t)blockIdx.x * blockDim.x + threadIdx.x; i < n; i += stride)
    p[i] = 0.0f;
}

__global__ void k_count(const int* __restrict__ idx, float* __restrict__ deg, int E) {
  int stride = gridDim.x * blockDim.x;
  for (int e = blockIdx.x * blockDim.x + threadIdx.x; e < E; e += stride)
    atomicAdd(&deg[idx[e]], 1.0f);
}

__global__ void k_rsqrt1(float* d, int n) {
  int i = blockIdx.x * blockDim.x + threadIdx.x;
  if (i < n) d[i] = rsqrtf(d[i] + 1.0f);
}

// W[K][N] f32 -> Wt[N][K] bf16
__global__ void k_transpose(const float* __restrict__ W, __bf16* __restrict__ Wt, int K, int N) {
  size_t tot = (size_t)K * N;
  size_t stride = (size_t)gridDim.x * blockDim.x;
  for (size_t i = (size_t)blockIdx.x * blockDim.x + threadIdx.x; i < tot; i += stride) {
    int k = (int)(i / N), n = (int)(i % N);
    Wt[(size_t)n * K + k] = f2bf(W[i]);
  }
}

__global__ void k_embed(const int* __restrict__ x, const float* __restrict__ emb,
                        __bf16* __restrict__ catbuf) {
  int n = blockIdx.x, c = threadIdx.x;
  catbuf[(size_t)n * CAT + c] = f2bf(emb[(size_t)x[n] * C + c]);
}

// ---------------- WMMA GEMM: out = [relu](A[N x K] @ Wt^T + bias) ----------------
// A: bf16 row-major, stride lda.  Wt: bf16 [C][K] (pre-transposed weight).
// Register-blocked: each wave computes 32 rows x 64 cols (2x4 tiles, 8 accumulators),
// 8 waves/block.  Per k-step: 12 x b128 loads feed 8 v_wmma_f32_16x16x32_bf16.
__global__ void __launch_bounds__(256)
k_gemm(const __bf16* __restrict__ A, int lda, int K,
       const __bf16* __restrict__ Wt, const float* __restrict__ bias,
       float* __restrict__ outF, __bf16* __restrict__ outB, int ldo,
       int nRows, int doRelu) {
  const int lane = threadIdx.x & 31;
  const int wave = threadIdx.x >> 5;
  const int m0   = (blockIdx.x * 8 + wave) * 32;
  if (m0 >= nRows) return;                        // wave-uniform
  const bool twoRows = (m0 + 16) < nRows;         // wave-uniform (3125 tiles is odd)
  const int n0   = blockIdx.y * 64;
  const int half = lane >> 4;
  const int l16  = lane & 15;

  const __bf16* pa0 = A + (size_t)(m0 + l16) * lda + half * 8;
  const __bf16* pa1 = pa0 + (size_t)16 * lda;
  const __bf16* pb  = Wt + (size_t)(n0 + l16) * K + half * 8;
  const size_t bstr = (size_t)16 * K;             // next 16-col tile of Wt

  v8f acc0[4] = {};
  v8f acc1[4] = {};

  for (int k = 0; k < K; k += 32) {
    v16bf b[4];
    #pragma unroll
    for (int j = 0; j < 4; ++j) b[j] = ld_bfrag(pb + (size_t)j * bstr + k);
    v16bf a0 = ld_afrag(pa0 + k);
    #pragma unroll
    for (int j = 0; j < 4; ++j)
      acc0[j] = __builtin_amdgcn_wmma_f32_16x16x32_bf16(false, a0, false, b[j],
                                                        (short)0, acc0[j], false, false);
    if (twoRows) {
      v16bf a1 = ld_afrag(pa1 + k);
      #pragma unroll
      for (int j = 0; j < 4; ++j)
        acc1[j] = __builtin_amdgcn_wmma_f32_16x16x32_bf16(false, a1, false, b[j],
                                                          (short)0, acc1[j], false, false);
    }
  }

  #pragma unroll
  for (int j = 0; j < 4; ++j) {
    const int col = n0 + j * 16 + l16;
    const float bv = bias ? bias[col] : 0.0f;
    if (outF) {
      #pragma unroll
      for (int r = 0; r < 8; ++r) {
        float v = acc0[j][r] + bv;
        if (doRelu) v = fmaxf(v, 0.0f);
        outF[(size_t)(m0 + half * 8 + r) * ldo + col] = v;
      }
      if (twoRows) {
        #pragma unroll
        for (int r = 0; r < 8; ++r) {
          float v = acc1[j][r] + bv;
          if (doRelu) v = fmaxf(v, 0.0f);
          outF[(size_t)(m0 + 16 + half * 8 + r) * ldo + col] = v;
        }
      }
    } else {
      #pragma unroll
      for (int r = 0; r < 8; ++r) {
        float v = acc0[j][r] + bv;
        if (doRelu) v = fmaxf(v, 0.0f);
        outB[(size_t)(m0 + half * 8 + r) * ldo + col] = f2bf(v);
      }
      if (twoRows) {
        #pragma unroll
        for (int r = 0; r < 8; ++r) {
          float v = acc1[j][r] + bv;
          if (doRelu) v = fmaxf(v, 0.0f);
          outB[(size_t)(m0 + 16 + half * 8 + r) * ldo + col] = f2bf(v);
        }
      }
    }
  }
}

// ---------------- edge scatter: agg[dst] += t2[src] * dinv[src]*dinv[dst] ----------------
__global__ void k_scatter(const int* __restrict__ e, const float* __restrict__ dinv,
                          const float* __restrict__ t2, float* __restrict__ agg, int E) {
  size_t total  = (size_t)E * (C / 4);
  size_t stride = (size_t)gridDim.x * blockDim.x;
  for (size_t i = (size_t)blockIdx.x * blockDim.x + threadIdx.x; i < total; i += stride) {
    int eid = (int)(i >> 6);
    int q   = (int)(i & 63) << 2;
    int src = e[eid];
    int dst = e[E + eid];
    float norm = dinv[src] * dinv[dst];
    const float4 v = *(const float4*)(t2 + (size_t)src * C + q);
    float* o = agg + (size_t)dst * C + q;
    atomicAdd(o + 0, v.x * norm);
    atomicAdd(o + 1, v.y * norm);
    atomicAdd(o + 2, v.z * norm);
    atomicAdd(o + 3, v.w * norm);
  }
}

// out = relu(agg + t2*dinv^2 + bias) -> bf16 into catbuf column block
__global__ void k_finish(const float* __restrict__ agg, const float* __restrict__ t2,
                         const float* __restrict__ dinv, const float* __restrict__ bias,
                         __bf16* __restrict__ catbuf, int coloff) {
  size_t total  = (size_t)N_NODES * C;
  size_t stride = (size_t)gridDim.x * blockDim.x;
  for (size_t i = (size_t)blockIdx.x * blockDim.x + threadIdx.x; i < total; i += stride) {
    int n = (int)(i >> 8), c = (int)(i & 255);
    float dv = dinv[n];
    float v = agg[i] + t2[i] * dv * dv + bias[c];
    catbuf[(size_t)n * CAT + coloff + c] = f2bf(fmaxf(v, 0.0f));
  }
}

// column sums of catbuf (bf16) into gsum (f32), split over row chunks
__global__ void k_colmean(const __bf16* __restrict__ catbuf, float* __restrict__ gsum) {
  int col = blockIdx.x * 256 + threadIdx.x;
  int r0  = blockIdx.y * 1000;
  float s = 0.0f;
  for (int r = r0; r < r0 + 1000; ++r)
    s += bf2f(catbuf[(size_t)r * CAT + col]);
  atomicAdd(&gsum[col], s);
}

// tiny MLP head: hid = relu(g@hW+hb); out = [2*tanh(hid@mW+mb), 2*sigmoid(hid@vW+vb)]
__global__ void __launch_bounds__(128)
k_head(const float* __restrict__ gsum, const float* __restrict__ hW, const float* __restrict__ hb,
       const float* __restrict__ mW, const float* __restrict__ mb,
       const float* __restrict__ vW, const float* __restrict__ vb, float* __restrict__ out) {
  __shared__ float g[CAT];
  __shared__ float hid[HID];
  int t = threadIdx.x;
  const float inv_n = 1.0f / (float)N_NODES;
  for (int k = t; k < CAT; k += 128) g[k] = gsum[k] * inv_n;
  __syncthreads();
  float acc = hb[t];
  for (int k = 0; k < CAT; ++k) acc += g[k] * hW[(size_t)k * HID + t];
  hid[t] = fmaxf(acc, 0.0f);
  __syncthreads();
  if (t < NOUT) {
    float m = mb[t];
    for (int k = 0; k < HID; ++k) m += hid[k] * mW[(size_t)k * NOUT + t];
    out[t] = 2.0f * tanhf(m);
  } else {
    int j = t - NOUT;
    float v = vb[j];
    for (int k = 0; k < HID; ++k) v += hid[k] * vW[(size_t)k * NOUT + j];
    out[NOUT + j] = 2.0f / (1.0f + expf(-v));
  }
}

// ---------------- host orchestration ----------------

extern "C" void kernel_launch(void* const* d_in, const int* in_sizes, int n_in,
                              void* d_out, int out_size, void* d_ws, size_t ws_size,
                              hipStream_t stream) {
  (void)in_sizes; (void)n_in; (void)out_size; (void)ws_size;
  const int*   x   = (const int*)d_in[0];
  const int*   ei  = (const int*)d_in[1];
  const int*   eit = (const int*)d_in[2];
  const float* emb = (const float*)d_in[3];
  const float *linW[NL][2], *linB[NL][2], *convW[NL][2], *convB[NL][2];
  for (int l = 0; l < NL; ++l) {
    int b = 4 + l * 8;
    linW[l][0]  = (const float*)d_in[b + 0]; linB[l][0]  = (const float*)d_in[b + 1];
    linW[l][1]  = (const float*)d_in[b + 2]; linB[l][1]  = (const float*)d_in[b + 3];
    convW[l][0] = (const float*)d_in[b + 4]; convB[l][0] = (const float*)d_in[b + 5];
    convW[l][1] = (const float*)d_in[b + 6]; convB[l][1] = (const float*)d_in[b + 7];
  }
  const float* hW = (const float*)d_in[36]; const float* hb = (const float*)d_in[37];
  const float* mW = (const float*)d_in[38]; const float* mb = (const float*)d_in[39];
  const float* vW = (const float*)d_in[40]; const float* vb = (const float*)d_in[41];

  // carve workspace (each region 256B aligned)
  char* wsb = (char*)d_ws;
  size_t off = 0;
  auto carve = [&](size_t bytes) -> char* {
    char* p = wsb + off;
    off = (off + bytes + 255) & ~(size_t)255;
    return p;
  };
  __bf16* catbuf = (__bf16*)carve((size_t)N_NODES * CAT * 2);
  __bf16* tmp1   = (__bf16*)carve((size_t)N_NODES * C * 2);
  float*  tmp2   = (float*) carve((size_t)N_NODES * C * 4);
  float*  agg    = (float*) carve((size_t)N_NODES * C * 4);
  float*  dinvA  = (float*) carve((size_t)N_NODES * 4);
  float*  dinvB  = (float*) carve((size_t)N_NODES * 4);
  float*  gsum   = (float*) carve((size_t)CAT * 4);
  __bf16* wt_lin[NL][2];
  __bf16* wt_conv[NL][2];
  for (int l = 0; l < NL; ++l) {
    size_t fin = (size_t)(2 * l + 1) * C;
    for (int d = 0; d < 2; ++d) {
      wt_lin[l][d]  = (__bf16*)carve(fin * C * 2);
      wt_conv[l][d] = (__bf16*)carve((size_t)C * C * 2);
    }
  }

  // convert / transpose weights to bf16 [N][K]
  for (int l = 0; l < NL; ++l) {
    int fin = (2 * l + 1) * C;
    for (int d = 0; d < 2; ++d) {
      k_transpose<<<512, 256, 0, stream>>>(linW[l][d],  wt_lin[l][d],  fin, C);
      k_transpose<<<256, 256, 0, stream>>>(convW[l][d], wt_conv[l][d], C,   C);
    }
  }

  // degrees -> dinv (per direction, same for all layers)
  k_zero<<<196, 256, 0, stream>>>(dinvA, N_NODES);
  k_zero<<<196, 256, 0, stream>>>(dinvB, N_NODES);
  k_count<<<1024, 256, 0, stream>>>(ei  + N_EDGES, dinvA, N_EDGES);
  k_count<<<1024, 256, 0, stream>>>(eit + N_EDGES, dinvB, N_EDGES);
  k_rsqrt1<<<(N_NODES + 255) / 256, 256, 0, stream>>>(dinvA, N_NODES);
  k_rsqrt1<<<(N_NODES + 255) / 256, 256, 0, stream>>>(dinvB, N_NODES);

  // embedding lookup -> first 256 columns of catbuf
  k_embed<<<N_NODES, C, 0, stream>>>(x, emb, catbuf);

  const int rowJobs = (N_NODES + 31) / 32;        // 32 rows per wave
  dim3 gemmGrid((rowJobs + 7) / 8, C / 64);       // 196 x 4

  for (int l = 0; l < NL; ++l) {
    int fin = (2 * l + 1) * C;
    for (int d = 0; d < 2; ++d) {
      // h = relu(cat @ lin_W + lin_b)  -> tmp1 (bf16)
      k_gemm<<<gemmGrid, 256, 0, stream>>>(catbuf, CAT, fin, wt_lin[l][d], linB[l][d],
                                           nullptr, tmp1, C, N_NODES, 1);
      // t2 = h @ conv_W  -> tmp2 (f32)
      k_gemm<<<gemmGrid, 256, 0, stream>>>(tmp1, C, C, wt_conv[l][d], nullptr,
                                           tmp2, nullptr, C, N_NODES, 0);
      // aggregation
      k_zero<<<4096, 256, 0, stream>>>(agg, (size_t)N_NODES * C);
      const int*   e    = (d == 0) ? ei    : eit;
      const float* dinv = (d == 0) ? dinvA : dinvB;
      k_scatter<<<8192, 256, 0, stream>>>(e, dinv, tmp2, agg, N_EDGES);
      // relu(agg + self + bias) appended into catbuf
      int coloff = (2 * l + 1 + d) * C;
      k_finish<<<4096, 256, 0, stream>>>(agg, tmp2, dinv, convB[l][d], catbuf, coloff);
    }
  }

  // graph readout mean + head MLP
  k_zero<<<9, 256, 0, stream>>>(gsum, (size_t)CAT);
  dim3 cmGrid(CAT / 256, 50);
  k_colmean<<<cmGrid, 256, 0, stream>>>(catbuf, gsum);
  k_head<<<1, 128, 0, stream>>>(gsum, hW, hb, mW, mb, vW, vb, (float*)d_out);
}
